// GraphAttentionLayer_33011118637077
// MI455X (gfx1250) — matrix operand order repaired and checked by reading, not compiled, not run on hardware
//
#include <hip/hip_runtime.h>

// GAT layer for MI455X (gfx1250, wave32, WMMA + TDM).
// Phase 1: Wh = h@W (fp32), stored f16 pre-swizzled into WMMA B-matrix lane
//          layout; per-row scalars Wh1 = Wh@a[:128], Wh2 = Wh@a[128:].
// Phase 1b: global max of Wh2 (softmax stabilizer upper bound).
// Phase 2: streaming masked softmax + attention@Wh via v_wmma_f32_16x16x32_f16.
//          Row sums of P computed on the matrix unit via a ones-B WMMA tile.
//          B tiles staged through LDS with the Tensor Data Mover; the 16
//          ds_load_b128 are issued via inline asm into 16 distinct register
//          quads, LDS latency is hidden behind the score/exp VALU work, and a
//          single s_wait_dscnt 0 precedes the 9-WMMA burst.

#define N_NODES 12288
#define IN_F    256
#define OUT_F   128
#define LALPHA  0.2f
#define NKB     (N_NODES / 32)   // 384 key blocks
#define KB_BYTES 8192            // one key block of swizzled Wh f16

typedef __attribute__((ext_vector_type(16))) _Float16 v16h;
typedef __attribute__((ext_vector_type(8)))  float    v8f;
typedef unsigned int u32;
typedef __attribute__((ext_vector_type(4))) u32 u32x4;
typedef __attribute__((ext_vector_type(8))) int i32x8;
typedef __attribute__((ext_vector_type(4))) int i32x4;

union BCast { float4 f[2]; v16h h; };

__device__ __forceinline__ v16h mkB(float4 lo, float4 hi) {
  BCast u; u.f[0] = lo; u.f[1] = hi; return u.h;
}

#if __has_builtin(__builtin_amdgcn_tensor_load_to_lds) && \
    __has_builtin(__builtin_amdgcn_s_wait_tensorcnt)
#define HAVE_TDM 1
#else
#define HAVE_TDM 0
#endif

#if HAVE_TDM
// Raw ds_load_b128 with a literal offset: forces a distinct destination quad
// per load so all 16 loads pipeline; DScnt is drained manually (see below).
#define DSLOAD(dst, addr, OFFSTR)                                        \
  asm volatile("ds_load_b128 %0, %1 offset:" OFFSTR                     \
               : "=v"(dst) : "v"(addr))

// 1-D TDM copy of 8KB (1024 x 8-byte elements) global -> LDS.
// D# built per CDNA5 ISA 8.3/8.4: group0 = {count, lds_addr, global_addr, type=2},
// group1 = {data_size=8B, tensor_dim0=tile_dim0=1024, tensor_dim1=tile_dim1=1}.
__device__ __forceinline__ void tdm_load_8k(u32 lds_off, const void* gptr) {
  const unsigned long long ga = (unsigned long long)gptr;
  u32x4 g0;
  g0[0] = 1u;                                           // count=1, user mode
  g0[1] = lds_off;                                      // LDS byte address
  g0[2] = (u32)(ga & 0xFFFFFFFFu);                      // global_addr[31:0]
  g0[3] = (u32)((ga >> 32) & 0x1FFFFFFu) | (2u << 30);  // [56:32] | type=2
  i32x8 g1;
  g1[0] = (int)(3u << 16);      // workgroup_mask=0, data_size=3 (8 bytes)
  g1[1] = (int)(1024u << 16);   // tensor_dim0[15:0]=1024
  g1[2] = (int)(1u << 16);      // tensor_dim0[31:16]=0 | tensor_dim1=1
  g1[3] = (int)(1024u << 16);   // tensor_dim1 hi=0 | tile_dim0=1024
  g1[4] = 1;                    // tile_dim1=1, tile_dim2=0
  g1[5] = 1024;                 // tensor_dim0_stride (don't-care for 1 row)
  g1[6] = 0;
  g1[7] = 0;
  const i32x4 z4 = {0, 0, 0, 0};
  const i32x8 z8 = {0, 0, 0, 0, 0, 0, 0, 0};
  __builtin_amdgcn_tensor_load_to_lds(g0, g1, z4, z4, z8, 0);
}
#endif

// ---------------- Phase 1: Wh, Wh1, Wh2, swizzled B-tiles -------------------
__global__ __launch_bounds__(128)
void gat_phase1(const float* __restrict__ h, const float* __restrict__ W,
                const float* __restrict__ a, float* __restrict__ Wh1,
                float* __restrict__ Wh2, _Float16* __restrict__ wsB) {
  __shared__ float hs[IN_F];
  __shared__ float r1[4], r2[4];
  const int i = blockIdx.x;          // node row
  const int o = threadIdx.x;         // output feature 0..127
  hs[o]       = h[(size_t)i * IN_F + o];
  hs[o + 128] = h[(size_t)i * IN_F + o + 128];
  __syncthreads();

  float acc = 0.f;
#pragma unroll 8
  for (int k = 0; k < IN_F; ++k)
    acc = fmaf(hs[k], W[k * OUT_F + o], acc);

  // Store Wh[i][o] as f16 directly in WMMA B-matrix (32x16 f16) lane layout:
  // key block jb = i/32, kin = i%32; col block nb = o/16, n = o%16.
  // lane = n + 16*((kin/8)%2); halfword t = kin%8 + 8*(kin/16).
  {
    const int jb = i >> 5, kin = i & 31;
    const int nb = o >> 4, nloc = o & 15;
    const int L = nloc + (((kin >> 3) & 1) << 4);
    const int t = (kin & 7) + ((kin >> 4) << 3);
    wsB[((size_t)(jb * 8 + nb) * 32 + L) * 16 + t] = (_Float16)acc;
  }

  // Row reductions: Wh1[i] = sum_o Wh[i][o]*a[o], Wh2[i] likewise
  float p1 = acc * a[o];
  float p2 = acc * a[OUT_F + o];
#pragma unroll
  for (int off = 16; off > 0; off >>= 1) {
    p1 += __shfl_down(p1, off, 32);
    p2 += __shfl_down(p2, off, 32);
  }
  const int lane = threadIdx.x & 31, wv = threadIdx.x >> 5;
  if (lane == 0) { r1[wv] = p1; r2[wv] = p2; }
  __syncthreads();
  if (threadIdx.x == 0) {
    Wh1[i] = r1[0] + r1[1] + r1[2] + r1[3];
    Wh2[i] = r2[0] + r2[1] + r2[2] + r2[3];
  }
}

// ---------------- Phase 1b: max(Wh2) ---------------------------------------
__global__ __launch_bounds__(256)
void gat_rowmax(const float* __restrict__ Wh2, float* __restrict__ M2, int n) {
  __shared__ float red[256];
  float mx = -3.4e38f;
  for (int i = threadIdx.x; i < n; i += 256) mx = fmaxf(mx, Wh2[i]);
  red[threadIdx.x] = mx;
  __syncthreads();
  for (int s = 128; s > 0; s >>= 1) {
    if (threadIdx.x < s)
      red[threadIdx.x] = fmaxf(red[threadIdx.x], red[threadIdx.x + s]);
    __syncthreads();
  }
  if (threadIdx.x == 0) *M2 = red[0];
}

// ---------------- Phase 2: masked softmax + attention@Wh via WMMA -----------
__global__ __launch_bounds__(128)
void gat_phase2(const int* __restrict__ adj, const float* __restrict__ Wh1,
                const float* __restrict__ Wh2, const float* __restrict__ M2,
                const _Float16* __restrict__ wsB, float* __restrict__ out) {
  const int lane  = threadIdx.x & 31;
  const int wv    = threadIdx.x >> 5;
  const int qt    = blockIdx.x * 4 + wv;   // query tile (16 rows), one per wave
  const int q0    = qt << 4;
  const int m     = lane & 15;             // my query row within tile
  const int half  = lane >> 4;             // 0: lanes 0-15, 1: lanes 16-31
  const int kbase = half << 3;             // A-layout K base (0 or 8)

  const float w1 = Wh1[q0 + m];
  float sm = w1 + *M2;                     // upper bound of row max (leaky mono)
  sm = sm > 0.f ? sm : LALPHA * sm;

  v8f acc[8] = {};                         // 8 col tiles x 16x16 f32 C
  v8f accS   = {};                         // row-sum tile: P x ones(32x16)
  v16h Bones;
#pragma unroll
  for (int t = 0; t < 16; ++t) Bones[t] = (_Float16)1.0f;

  const int4*   arow = (const int4*)(adj + (size_t)(q0 + m) * N_NODES);
  const float4* w2v  = (const float4*)Wh2;

#if HAVE_TDM
  __shared__ _Float16 Bbuf[2][KB_BYTES / 2];   // double-buffered key-block tiles
  if (wv == 0)
    tdm_load_8k((u32)(uintptr_t)&Bbuf[0][0], wsB);       // prologue: jb=0
#else
  const float4* bv = (const float4*)wsB;
#endif

  for (int jb = 0; jb < NKB; ++jb) {
    const int j0 = jb << 5;
    const int b4 = (j0 + kbase) >> 2;
    __builtin_prefetch(arow + b4 + 8, 0, 3);   // next key block's adj rows

    const int4   a0 = arow[b4], a1 = arow[b4 + 1];
    const int4   a2 = arow[b4 + 4], a3 = arow[b4 + 5];
    const float4 u0 = w2v[b4],  u1 = w2v[b4 + 1];
    const float4 u2 = w2v[b4 + 4], u3 = w2v[b4 + 5];

    const int   av[16]  = {a0.x, a0.y, a0.z, a0.w, a1.x, a1.y, a1.z, a1.w,
                           a2.x, a2.y, a2.z, a2.w, a3.x, a3.y, a3.z, a3.w};
    const float wv2[16] = {u0.x, u0.y, u0.z, u0.w, u1.x, u1.y, u1.z, u1.w,
                           u2.x, u2.y, u2.z, u2.w, u3.x, u3.y, u3.z, u3.w};

#if HAVE_TDM
    const int cur = jb & 1;
    if (wv == 0) __builtin_amdgcn_s_wait_tensorcnt((short)0);
    __syncthreads();                        // tile[cur] ready; tile[cur^1] free
    if (wv == 0 && (jb + 1) < NKB)
      tdm_load_8k((u32)(uintptr_t)&Bbuf[cur ^ 1][0],
                  (const char*)wsB + (size_t)(jb + 1) * KB_BYTES);

    // Issue all 16 ds_load_b128 into DISTINCT quads (asm outputs cannot be
    // coalesced by RA).  DScnt is drained once, after the A-tile VALU work.
    const u32 lb = (u32)(uintptr_t)&Bbuf[cur][0] + (u32)lane * 32u;
    float4 c0, c1, c2, c3, c4, c5, c6, c7;
    float4 c8, c9, c10, c11, c12, c13, c14, c15;
    DSLOAD(c0,  lb, "0");    DSLOAD(c1,  lb, "16");
    DSLOAD(c2,  lb, "1024"); DSLOAD(c3,  lb, "1040");
    DSLOAD(c4,  lb, "2048"); DSLOAD(c5,  lb, "2064");
    DSLOAD(c6,  lb, "3072"); DSLOAD(c7,  lb, "3088");
    DSLOAD(c8,  lb, "4096"); DSLOAD(c9,  lb, "4112");
    DSLOAD(c10, lb, "5120"); DSLOAD(c11, lb, "5136");
    DSLOAD(c12, lb, "6144"); DSLOAD(c13, lb, "6160");
    DSLOAD(c14, lb, "7168"); DSLOAD(c15, lb, "7184");
#endif

    // Build A tile in exact 16-bit A-matrix layout: halfword t holds
    // K = kbase + t (+8 for t>=8).  This VALU/exp work (~16 transcendentals)
    // executes while the 16 LDS loads are in flight.
    v16h A;
#pragma unroll
    for (int t = 0; t < 16; ++t) {
      float s = w1 + wv2[t];
      s = s > 0.f ? s : LALPHA * s;                        // LeakyReLU
      const float p = (av[t] > 0) ? __expf(s - sm) : 0.f;  // mask -> 0
      A[t] = (_Float16)p;
    }

#if HAVE_TDM
    asm volatile("s_wait_dscnt 0x0" ::: "memory");   // drain the 16 asm loads
    v16h Bt[8] = {mkB(c0, c1),   mkB(c2, c3),   mkB(c4, c5),   mkB(c6, c7),
                  mkB(c8, c9),   mkB(c10, c11), mkB(c12, c13), mkB(c14, c15)};
#else
    v16h Bt[8];
#pragma unroll
    for (int nb = 0; nb < 8; ++nb) {
      BCast B;
      const float4* bp = bv + (((size_t)(jb * 8 + nb) * 32 + lane) << 1);
      B.f[0] = bp[0];
      B.f[1] = bp[1];
      Bt[nb] = B.h;
    }
#endif

#pragma unroll
    for (int nb = 0; nb < 8; ++nb)
      acc[nb] = __builtin_amdgcn_wmma_f32_16x16x32_f16(
          false, A, false, Bt[nb], (short)0, acc[nb], false, false);

    // Row sums on the matrix unit: every column of accS = sum_k P[m][k]
    accS = __builtin_amdgcn_wmma_f32_16x16x32_f16(
        false, A, false, Bones, (short)0, accS, false, false);
  }

  // C layout: VGPR r, lanes 0-15 -> M=r, lanes 16-31 -> M=r+8; N = lane%16.
  // accS[r] in this lane is exactly the softmax denominator of row q0+r+8*half.
#pragma unroll
  for (int r = 0; r < 8; ++r) {
    const float l   = accS[r];
    const float inv = (l > 0.f) ? 1.f / l : 0.f;
    const int   row = q0 + r + (half << 3);
#pragma unroll
    for (int nb = 0; nb < 8; ++nb) {
      float v = acc[nb][r] * inv;
      v = v > 0.f ? v : expm1f(v);                 // ELU (alpha = 1)
      out[(size_t)row * OUT_F + nb * 16 + m] = v;
    }
  }
}

// ---------------- Launch ----------------------------------------------------
extern "C" void kernel_launch(void* const* d_in, const int* in_sizes, int n_in,
                              void* d_out, int out_size, void* d_ws, size_t ws_size,
                              hipStream_t stream) {
  const float* h   = (const float*)d_in[0];
  const int*   adj = (const int*)d_in[1];
  const float* W   = (const float*)d_in[2];
  const float* a   = (const float*)d_in[3];
  float*       out = (float*)d_out;

  char* ws = (char*)d_ws;
  _Float16* wsB = (_Float16*)ws;                                   // 3 MB
  float* Wh1 = (float*)(ws + (size_t)N_NODES * OUT_F * sizeof(_Float16));
  float* Wh2 = Wh1 + N_NODES;
  float* M2  = Wh2 + N_NODES;

  gat_phase1<<<N_NODES, 128, 0, stream>>>(h, W, a, Wh1, Wh2, wsB);
  gat_rowmax<<<1, 256, 0, stream>>>(Wh2, M2, N_NODES);
  gat_phase2<<<N_NODES / 64, 128, 0, stream>>>(adj, Wh1, Wh2, M2, wsB, out);
}
